// TransformerBlockQuantum_65481071403191
// MI455X (gfx1250) — matrix operand order repaired and checked by reading, not compiled, not run on hardware
//
#include <hip/hip_runtime.h>

typedef __attribute__((ext_vector_type(16))) _Float16 v16h;
typedef __attribute__((ext_vector_type(8)))  _Float16 v8h;
typedef __attribute__((ext_vector_type(4)))  _Float16 v4h;
typedef __attribute__((ext_vector_type(8)))  float    v8f;
typedef __attribute__((ext_vector_type(4)))  float    v4f;

#define E_DIM 1024
#define NQ_DIM 64

// ---------------------------------------------------------------------------
// z[i] = cos(x[i] + a[i%64]) * cos(b[i%64])  -> f16
// ---------------------------------------------------------------------------
__global__ void k_prep_z(const float* __restrict__ x,
                         const float* __restrict__ qp,   // (64,3) row-major
                         _Float16* __restrict__ zh, int n) {
  int i = blockIdx.x * blockDim.x + threadIdx.x;
  if (i >= n) return;
  int d = i & 63;
  float a = qp[d * 3 + 0];
  float b = qp[d * 3 + 1];
  zh[i] = (_Float16)(__cosf(x[i] + a) * __cosf(b));
}

// ---------------------------------------------------------------------------
// f32 -> f16 convert
// ---------------------------------------------------------------------------
__global__ void k_cvt_f16(const float* __restrict__ in,
                          _Float16* __restrict__ out, int n) {
  int i = blockIdx.x * blockDim.x + threadIdx.x;
  if (i >= n) return;
  out[i] = (_Float16)in[i];
}

// ---------------------------------------------------------------------------
// out[M,N] = A[M,K](f16) @ Bw[N,K](f16)^T + resid[M,N] + bias[N]
// Workgroup: 256 threads (8 wave32), tile 128(M) x 128(N).
// Wave (wid>>2 -> M half, wid&3 -> N quarter) computes 64x32 = 4x2 fragments.
// ---------------------------------------------------------------------------
__global__ __launch_bounds__(256)
void k_gemm_wmma(const _Float16* __restrict__ A,
                 const _Float16* __restrict__ Bw,
                 const float* __restrict__ resid,
                 const float* __restrict__ bias,
                 float* __restrict__ out,
                 int M, int N, int K) {
  const int lane = threadIdx.x & 31;
  const int wid  = threadIdx.x >> 5;
  const int m0 = blockIdx.y * 128 + (wid >> 2) * 64;
  const int n0 = blockIdx.x * 128 + (wid & 3) * 32;
  const int rsel = lane & 15;       // row-within-fragment selector
  const int hi   = lane >> 4;       // lane-group (K-half) selector

  v8f zero = {};
  v8f acc[4][2];
#pragma unroll
  for (int i = 0; i < 4; ++i)
#pragma unroll
    for (int j = 0; j < 2; ++j) acc[i][j] = zero;

  // A: lane (rsel, hi) holds K chunks {hi*8 .. hi*8+7} and {16+hi*8 ..}
  const _Float16* Abase = A + (size_t)(m0 + rsel) * K + hi * 8;
  // B: lane (rsel, hi) holds K chunks {hi*8 .. hi*8+7} and {16+hi*8 ..} of column rsel
  const _Float16* Bbase = Bw + (size_t)(n0 + rsel) * K + hi * 8;

#pragma unroll 2
  for (int kk = 0; kk < K; kk += 32) {
    v16h a[4], b[2];
#pragma unroll
    for (int i = 0; i < 4; ++i) {
      const _Float16* p = Abase + (size_t)(16 * i) * K + kk;
      v8h* ah = reinterpret_cast<v8h*>(&a[i]);
      ah[0] = *(const v8h*)(p);
      ah[1] = *(const v8h*)(p + 16);
    }
#pragma unroll
    for (int j = 0; j < 2; ++j) {
      const _Float16* p = Bbase + (size_t)(16 * j) * K + kk;
      v8h* bh = reinterpret_cast<v8h*>(&b[j]);
      bh[0] = *(const v8h*)(p);
      bh[1] = *(const v8h*)(p + 16);
    }
#pragma unroll
    for (int i = 0; i < 4; ++i)
#pragma unroll
      for (int j = 0; j < 2; ++j)
        acc[i][j] = __builtin_amdgcn_wmma_f32_16x16x32_f16(
            false, a[i], false, b[j], (short)0, acc[i][j], false, false);
  }

  // Epilogue: C/D layout -> col = n0+16j+rsel, row = m0+16i + v + hi*8
#pragma unroll
  for (int j = 0; j < 2; ++j) {
    int col = n0 + 16 * j + rsel;
    float bs = bias[col];
#pragma unroll
    for (int i = 0; i < 4; ++i) {
#pragma unroll
      for (int v = 0; v < 8; ++v) {
        int row = m0 + 16 * i + v + hi * 8;
        size_t idx = (size_t)row * N + col;
        out[idx] = acc[i][j][v] + resid[idx] + bs;
      }
    }
  }
}

// ---------------------------------------------------------------------------
// Row-wise LayerNorm over 1024 cols; one wave32 per row, 32 floats/lane.
// Optionally emits f16 quantum features q = cos(x1[:, :64]+fa)*cos(fb).
// ---------------------------------------------------------------------------
__global__ __launch_bounds__(256)
void k_layernorm(const float* __restrict__ y,
                 const float* __restrict__ w,
                 const float* __restrict__ b,
                 float* __restrict__ out,
                 _Float16* __restrict__ qout,        // may be nullptr
                 const float* __restrict__ qp,       // (64,3) or nullptr
                 int rows) {
  int gw = (blockIdx.x * blockDim.x + threadIdx.x) >> 5;
  int lane = threadIdx.x & 31;
  if (gw >= rows) return;
  const float* row = y + (size_t)gw * E_DIM;

  v4f vals[8];
  float s = 0.f, sq = 0.f;
#pragma unroll
  for (int i = 0; i < 8; ++i) {
    vals[i] = *(const v4f*)(row + i * 128 + lane * 4);
#pragma unroll
    for (int j = 0; j < 4; ++j) {
      s  += vals[i][j];
      sq += vals[i][j] * vals[i][j];
    }
  }
#pragma unroll
  for (int m = 16; m >= 1; m >>= 1) {
    s  += __shfl_xor(s, m, 32);
    sq += __shfl_xor(sq, m, 32);
  }
  const float inv = 1.0f / (float)E_DIM;
  float mu  = s * inv;
  float var = sq * inv - mu * mu;
  float rs  = __frsqrt_rn(var + 1e-5f);

#pragma unroll
  for (int i = 0; i < 8; ++i) {
    int col0 = i * 128 + lane * 4;
    v4f o;
#pragma unroll
    for (int j = 0; j < 4; ++j)
      o[j] = (vals[i][j] - mu) * rs * w[col0 + j] + b[col0 + j];
    *(v4f*)(out + (size_t)gw * E_DIM + col0) = o;
    if (i == 0 && qout != nullptr && lane < 16) {
      v4h q;
#pragma unroll
      for (int j = 0; j < 4; ++j) {
        int d = lane * 4 + j;
        q[j] = (_Float16)(__cosf(o[j] + qp[d * 3 + 0]) * __cosf(qp[d * 3 + 1]));
      }
      *(v4h*)(qout + (size_t)gw * NQ_DIM + lane * 4) = q;
    }
  }
}

// ---------------------------------------------------------------------------
extern "C" void kernel_launch(void* const* d_in, const int* in_sizes, int n_in,
                              void* d_out, int out_size, void* d_ws, size_t ws_size,
                              hipStream_t stream) {
  const float* x       = (const float*)d_in[0];
  const float* attn_qp = (const float*)d_in[1];
  const float* Wo      = (const float*)d_in[2];
  const float* bo      = (const float*)d_in[3];
  const float* ffn_qp  = (const float*)d_in[4];
  const float* Lw      = (const float*)d_in[5];
  const float* Lb      = (const float*)d_in[6];
  const float* n1w     = (const float*)d_in[7];
  const float* n1b     = (const float*)d_in[8];
  const float* n2w     = (const float*)d_in[9];
  const float* n2b     = (const float*)d_in[10];

  const int E  = E_DIM;            // 1024
  const int NQ = NQ_DIM;           // 64
  const int ME = in_sizes[0];      // B*S*E = 16,777,216
  const int M  = ME / E;           // 16384

  // Workspace layout (all offsets 16B+ aligned)
  char* ws = (char*)d_ws;
  float*    y  = (float*)   (ws);                                   // M*E f32
  float*    x1 = (float*)   (ws + (size_t)ME * 4);                  // M*E f32
  _Float16* zh = (_Float16*)(ws + (size_t)ME * 8);                  // M*E f16
  _Float16* Wh = (_Float16*)(ws + (size_t)ME * 10);                 // E*E f16
  _Float16* Lh = (_Float16*)(ws + (size_t)ME * 10 + (size_t)E*E*2); // E*NQ f16
  _Float16* qh = (_Float16*)(ws + (size_t)ME * 10 + (size_t)E*E*2
                                + (size_t)E*NQ*2);                  // M*NQ f16

  // 1) quantum attention features -> f16
  k_prep_z<<<(ME + 255) / 256, 256, 0, stream>>>(x, attn_qp, zh, ME);
  // 2) weight conversions -> f16
  k_cvt_f16<<<(E * E + 255) / 256, 256, 0, stream>>>(Wo, Wh, E * E);
  k_cvt_f16<<<(E * NQ + 255) / 256, 256, 0, stream>>>(Lw, Lh, E * NQ);

  dim3 gemm_grid(E / 128, M / 128);
  // 3) attn GEMM + residual: y = zh @ Wh^T + x + bo
  k_gemm_wmma<<<gemm_grid, 256, 0, stream>>>(zh, Wh, x, bo, y, M, E, E);
  // 4) LN1 -> x1, fused ffn quantum features -> qh
  k_layernorm<<<(M * 32 + 255) / 256, 256, 0, stream>>>(y, n1w, n1b, x1, qh,
                                                        ffn_qp, M);
  // 5) ffn GEMM + residual: y = qh @ Lh^T + x1 + Lb   (K = 64)
  k_gemm_wmma<<<gemm_grid, 256, 0, stream>>>(qh, Lh, x1, Lb, y, M, E, NQ);
  // 6) LN2 -> output
  k_layernorm<<<(M * 32 + 255) / 256, 256, 0, stream>>>(y, n2w, n2b,
                                                        (float*)d_out, nullptr,
                                                        nullptr, M);
}
